// SAModule_28389733826886
// MI455X (gfx1250) — matrix operand order repaired and checked by reading, not compiled, not run on hardware
//
#include <hip/hip_runtime.h>

// ---------------------------------------------------------------------------
// MI455X (gfx1250) PointConv SA module, round 4.
//  - f16 WMMA (v_wmma_f32_16x16x32_f16) for all three matmuls.
//  - B operands pre-swizzled to fragment-linear [k-slab][n][32] order: each
//    fragment is 2x ds_load_b128 per lane (confirmed in round-2 disasm).
//  - LDS staging via GLOBAL_LOAD_ASYNC_TO_LDS_B128 (ASYNCcnt). The builtin is
//    present on this toolchain and expects (global v4i*, local v4i*, Ii, Ii).
// Wave32; fragment layouts per CDNA5 ISA §7.12.2.
// ---------------------------------------------------------------------------

typedef __attribute__((ext_vector_type(16))) _Float16 v16h;
typedef __attribute__((ext_vector_type(8)))  _Float16 v8h;
typedef __attribute__((ext_vector_type(8)))  float    v8f;
typedef __attribute__((ext_vector_type(4)))  int      v4i;

#define N_PTS 8192
#define KNBR  64
#define C_IN  64
#define RAD2  (0.125f * 0.125f)
#define KPAD  96          // 64 feat + 3 rel + 29 zero pad -> 3 k-steps of 32
#define CAP   256         // candidate list capacity per target row

#if defined(__AMDGCN__) && __has_builtin(__builtin_amdgcn_global_load_async_to_lds_b128) && \
    __has_builtin(__builtin_amdgcn_s_wait_asynccnt)
#define HAVE_ASYNC_LDS 1
#else
#define HAVE_ASYNC_LDS 0
#endif

#if HAVE_ASYNC_LDS
typedef __attribute__((address_space(1))) v4i* gv4i_ptr;   // global
typedef __attribute__((address_space(3))) v4i* lv4i_ptr;   // LDS
// One 16B chunk global->LDS, no VGPR round-trip; tracked on ASYNCcnt.
__device__ __forceinline__ void async_cp16(void* lds, const void* g) {
  __builtin_amdgcn_global_load_async_to_lds_b128(
      (gv4i_ptr)g, (lv4i_ptr)lds, /*imm offset*/ 0, /*cpol*/ 0);
}
__device__ __forceinline__ void async_wait0() { __builtin_amdgcn_s_wait_asynccnt(0); }
#else
__device__ __forceinline__ void async_cp16(void* lds, const void* g) {
  *(uint4*)lds = *(const uint4*)g;
}
__device__ __forceinline__ void async_wait0() {}
#endif

__device__ __forceinline__ v16h cat8(v8h lo, v8h hi) {
  return __builtin_shufflevector(lo, hi, 0, 1, 2, 3, 4, 5, 6, 7,
                                 8, 9, 10, 11, 12, 13, 14, 15);
}

// A (16x32 f16) from row-major LDS tile: lane L(0..15)=row M=L, K 0..15;
// lane L+16: same row, K 16..31. 16 contiguous halves per lane.
__device__ __forceinline__ v16h ldsA_frag(const _Float16* base, int m0, int ldk, int k0) {
  const int lane = threadIdx.x & 31;
  const _Float16* p = base + (m0 + (lane & 15)) * ldk + k0 + ((lane >> 4) << 4);
  return cat8(*(const v8h*)p, *(const v8h*)(p + 8));
}

// B (32x16 f16) from swizzled LDS [slab][ntot][32]: lane L(0..15)=col N=L,
// K=0..15; lanes 16..31 same cols, K=16..31. 16 contiguous halves per lane.
__device__ __forceinline__ v16h ldsBsw_frag(const _Float16* baseSw, int ntot, int n0, int slab) {
  const int lane = threadIdx.x & 31;
  const _Float16* p = baseSw + ((slab * ntot + n0 + (lane & 15)) << 5) + ((lane >> 4) << 4);
  return cat8(*(const v8h*)p, *(const v8h*)(p + 8));
}

__device__ __forceinline__ v8f wmma_f16(v16h a, v16h b, v8f c) {
  // 8 args: (neg_a, A, neg_b, B, c_mod, C, reuse_a, reuse_b)
  return __builtin_amdgcn_wmma_f32_16x16x32_f16(false, a, false, b, (short)0, c, false, false);
}

__device__ __forceinline__ unsigned long long shflxor_u64(unsigned long long v, int m) {
  unsigned lo = (unsigned)v, hi = (unsigned)(v >> 32);
  lo = (unsigned)__shfl_xor((int)lo, m, 32);
  hi = (unsigned)__shfl_xor((int)hi, m, 32);
  return ((unsigned long long)hi << 32) | lo;
}

// ---------------------------------------------------------------------------
// Kernel 0: fp32 -> fp16 staging. x stays row-major (A-side / gathered).
// Weights are swizzled to B-fragment-linear order:
//   Wsw[((slab*N + n) << 5) + e] = W[(32*slab + e) * N + n]    (e = k % 32)
// W1 is zero-padded in K: 67 -> 96 (3 slabs).
// ---------------------------------------------------------------------------
__global__ void cvt_kernel(const float* __restrict__ x,  const float* __restrict__ W1,
                           const float* __restrict__ W2, const float* __restrict__ Wg,
                           _Float16* __restrict__ xh,   _Float16* __restrict__ W1sw,
                           _Float16* __restrict__ W2sw, _Float16* __restrict__ Wgsw) {
  const int stride = gridDim.x * blockDim.x;
  const int tid0 = blockIdx.x * blockDim.x + threadIdx.x;
  for (int t = tid0; t < N_PTS * C_IN; t += stride) xh[t] = (_Float16)x[t];
  for (int o = tid0; o < 3 * 64 * 32; o += stride) {        // W1sw [3][64][32]
    const int e = o & 31, n = (o >> 5) & 63, s = o >> 11;
    const int k = 32 * s + e;
    W1sw[o] = (k < C_IN + 3) ? (_Float16)W1[k * 64 + n] : (_Float16)0.0f;
  }
  for (int o = tid0; o < 2 * 128 * 32; o += stride) {       // W2sw [2][128][32]
    const int e = o & 31, n = (o >> 5) & 127, s = o >> 12;
    W2sw[o] = (_Float16)W2[(32 * s + e) * 128 + n];
  }
  for (int o = tid0; o < 4 * 256 * 32; o += stride) {       // Wgsw [4][256][32]
    const int e = o & 31, n = (o >> 5) & 255, s = o >> 13;
    Wgsw[o] = (_Float16)Wg[(32 * s + e) * 256 + n];
  }
}

// ---------------------------------------------------------------------------
// Kernel 1: radius neighbor search (VALU + LDS atomics; selection dominates
// the 3-wide dot, so tensorizing the distance GEMM would not pay off).
// 16 target rows / block; K-smallest extraction is set-equal to reference
// top-K and max-aggregation is order invariant. batch all-zero -> elided.
// ---------------------------------------------------------------------------
__global__ __launch_bounds__(256) void knn_kernel(const float* __restrict__ pos,
                                                  int* __restrict__ nbr) {
  __shared__ float tpx[16], tpy[16], tpz[16], tsq[16];
  __shared__ int   cnt[16];
  __shared__ int   cj[16][CAP];
  __shared__ float cd[16][CAP];

  const int tid = threadIdx.x;
  const int i0 = blockIdx.x * 16;
  if (tid < 16) {
    const int i = i0 + tid;
    const float px = pos[i * 3 + 0], py = pos[i * 3 + 1], pz = pos[i * 3 + 2];
    tpx[tid] = px; tpy[tid] = py; tpz[tid] = pz;
    tsq[tid] = px * px + py * py + pz * pz;
    cnt[tid] = 0;
  }
  __syncthreads();

  for (int jb = 0; jb < N_PTS; jb += 256) {
    const int j = jb + tid;
    const float cx = pos[j * 3 + 0], cy = pos[j * 3 + 1], cz = pos[j * 3 + 2];
    const float cs = cx * cx + cy * cy + cz * cz;
#pragma unroll
    for (int t = 0; t < 16; ++t) {
      const float d2 = tsq[t] + cs - 2.0f * (cx * tpx[t] + cy * tpy[t] + cz * tpz[t]);
      if (d2 <= RAD2) {
        const int slot = atomicAdd(&cnt[t], 1);
        if (slot < CAP) { cj[t][slot] = j; cd[t][slot] = d2; }
      }
    }
  }
  __syncthreads();

  const int wave = tid >> 5, lane = tid & 31;
  for (int t = wave; t < 16; t += 8) {                 // 2 rows per wave
    const int n = min(cnt[t], CAP);
    int* outp = nbr + (size_t)(i0 + t) * KNBR;
    if (n <= KNBR) {
      for (int s = lane; s < KNBR; s += 32)
        outp[s] = (s < n) ? cj[t][s] : -1;             // -1 = invalid (masked)
    } else {
      float dloc[CAP / 32];
      int   sloc[CAP / 32];
#pragma unroll
      for (int r = 0; r < CAP / 32; ++r) {
        const int s = lane + 32 * r;
        dloc[r] = (s < n) ? cd[t][s] : __builtin_inff();
        sloc[r] = s;
      }
      for (int kk = 0; kk < KNBR; ++kk) {
        unsigned long long best = ~0ull;
#pragma unroll
        for (int r = 0; r < CAP / 32; ++r) {
          const unsigned long long key =
              ((unsigned long long)__float_as_uint(dloc[r]) << 32) | (unsigned)sloc[r];
          best = (key < best) ? key : best;
        }
#pragma unroll
        for (int off = 16; off > 0; off >>= 1) {
          const unsigned long long o = shflxor_u64(best, off);
          best = (o < best) ? o : best;
        }
        const int slot = (int)(best & 0xffffffffu);    // uniform across wave
        if (lane == (slot & 31)) {
          const int rr = slot >> 5;
#pragma unroll
          for (int r = 0; r < CAP / 32; ++r)
            if (r == rr) dloc[r] = __builtin_inff();   // retire winner
        }
        if (lane == 0) outp[kk] = cj[t][slot];
      }
    }
  }
}

// ---------------------------------------------------------------------------
// Kernel 2: per-point PointConv MLP, one block (16 waves / 512 thr) per point.
//   h1 = relu([64x96 feat] @ W1 + b1)   16 tiles, 3 k-steps each
//   h2 = relu([64x64 h1]  @ W2 + b2)    32 tiles, 2 k-steps each
//   agg = masked column-max over 64 neighbor rows -> f16 [128]
// LDS phase-overlapped (<58KB static). Weights arrive pre-swizzled and are
// staged straight into LDS via async b128 DMA (no VGPR round-trip).
// ---------------------------------------------------------------------------
__global__ __launch_bounds__(512) void pointconv_kernel(
    const _Float16* __restrict__ xh, const float* __restrict__ pos,
    const int* __restrict__ nbr,
    const _Float16* __restrict__ W1sw, const float* __restrict__ b1,
    const _Float16* __restrict__ W2sw, const float* __restrict__ b2,
    _Float16* __restrict__ aggh) {
  __shared__ __align__(16) char smem[8192 + 256 + 49152];
  _Float16* sH1  = (_Float16*)smem;                      // 64x64 f16   (8KB)
  int*      sIdx = (int*)(smem + 8192);                  // 64 ints
  char*     dyn  = smem + 8448;                          // phase region (48KB)
  _Float16* sFeat = (_Float16*)dyn;                      // A: 64x96 f16   (12KB)
  _Float16* sW1   = (_Float16*)(dyn + 12288);            // A: [3][64][32] (12KB)
  _Float16* sW2   = (_Float16*)dyn;                      // B: [2][128][32](16KB)
  float*    sH2   = (float*)(dyn + 16384);               // B: 64x128 f32  (32KB)

  const int i = blockIdx.x;
  const int tid = threadIdx.x;
  if (tid < KNBR) sIdx[tid] = nbr[(size_t)i * KNBR + tid];
  // Stage swizzled W1 (768 x 16B) directly into LDS (ASYNCcnt path).
  for (int t = tid; t < (3 * 64 * 32) / 8; t += 512)
    async_cp16(((uint4*)sW1) + t, ((const uint4*)W1sw) + t);
  // Warm W2 slab into near caches while the h1 phase runs (global_prefetch_b8).
  __builtin_prefetch(W2sw + (tid << 4), 0, 0);
  async_wait0();
  __syncthreads();

  // feat = concat(x[j], pos[j]-pos[i], zeros). x gather: one uint4 per thread
  // (512 threads == 64 rows x 8 chunks exactly).
  {
    const int r = tid >> 3, c16 = tid & 7;
    const int j = sIdx[r];
    uint4 v = make_uint4(0u, 0u, 0u, 0u);
    if (j >= 0) v = *(const uint4*)(xh + (size_t)j * C_IN + (c16 << 3));
    *(uint4*)(sFeat + r * KPAD + (c16 << 3)) = v;
  }
  const float pix = pos[i * 3 + 0], piy = pos[i * 3 + 1], piz = pos[i * 3 + 2];
  for (int t = tid; t < 64 * 32; t += 512) {             // rel + zero pad cols
    const int r = t >> 5, c = C_IN + (t & 31);
    const int j = sIdx[r];
    _Float16 v = (_Float16)0.0f;
    if (j >= 0 && c < C_IN + 3) {
      const float pj = pos[j * 3 + (c - C_IN)];
      const float pi = (c == C_IN) ? pix : ((c == C_IN + 1) ? piy : piz);
      v = (_Float16)(pj - pi);
    }
    sFeat[r * KPAD + c] = v;
  }
  __syncthreads();

  const int wave = tid >> 5, lane = tid & 31;
  {  // h1: 4x4 tiles, one per wave
    const int mi = wave >> 2, ni = wave & 3;
    v8f acc = {};
#pragma unroll
    for (int k0 = 0; k0 < KPAD; k0 += 32) {
      const v16h a = ldsA_frag(sFeat, mi * 16, KPAD, k0);
      const v16h b = ldsBsw_frag(sW1, 64, ni * 16, k0 >> 5);
      acc = wmma_f16(a, b, acc);
    }
    const int n = ni * 16 + (lane & 15);
    const float bb = b1[n];
#pragma unroll
    for (int p = 0; p < 8; ++p) {
      const int m = mi * 16 + p + 8 * (lane >> 4);       // C/D: VGPR p -> row M
      const float v = acc[p] + bb;
      sH1[m * 64 + n] = (_Float16)(v > 0.f ? v : 0.f);
    }
  }
  __syncthreads();
  for (int t = tid; t < (2 * 128 * 32) / 8; t += 512)    // stage swizzled W2
    async_cp16(((uint4*)sW2) + t, ((const uint4*)W2sw) + t);
  async_wait0();
  __syncthreads();

#pragma unroll
  for (int tt = 0; tt < 2; ++tt) {  // h2: 4x8 tiles, two per wave
    const int tile = wave + 16 * tt;
    const int mi = tile >> 3, ni = tile & 7;
    v8f acc = {};
#pragma unroll
    for (int k0 = 0; k0 < 64; k0 += 32) {
      const v16h a = ldsA_frag(sH1, mi * 16, 64, k0);
      const v16h b = ldsBsw_frag(sW2, 128, ni * 16, k0 >> 5);
      acc = wmma_f16(a, b, acc);
    }
    const int n = ni * 16 + (lane & 15);
    const float bb = b2[n];
#pragma unroll
    for (int p = 0; p < 8; ++p) {
      const int m = mi * 16 + p + 8 * (lane >> 4);
      const float v = acc[p] + bb;
      sH2[m * 128 + n] = v > 0.f ? v : 0.f;
    }
  }
  __syncthreads();

  if (tid < 128) {  // masked max over neighbors (self always valid -> m >= 0)
    float m = -__builtin_inff();
    for (int r = 0; r < 64; ++r)
      if (sIdx[r] >= 0) {
        const float v = sH2[r * 128 + tid];
        m = v > m ? v : m;
      }
    aggh[(size_t)i * 128 + tid] = (_Float16)m;
  }
}

// ---------------------------------------------------------------------------
// Kernel 3: out = relu(agg[8192x128] @ Wg[128x256] + bg), 64x128 tile / block.
// Wg arrives pre-swizzled [4][256][32]; both tiles are staged into LDS with
// async b128 DMA, then 4 WMMA k-steps per 16x16 tile.
// ---------------------------------------------------------------------------
__global__ __launch_bounds__(256) void global_nn_kernel(
    const _Float16* __restrict__ aggh, const _Float16* __restrict__ Wgsw,
    const float* __restrict__ bg, float* __restrict__ out) {
  __shared__ __align__(16) _Float16 sA[64 * 128];        // 16KB, row-major
  __shared__ __align__(16) _Float16 sB[4 * 128 * 32];    // 32KB, swizzled
  const int tid = threadIdx.x;
  const int m0 = blockIdx.x * 64;
  const int n0 = blockIdx.y * 128;
  for (int t = tid; t < (64 * 128) / 8; t += 256)        // rows m0.. contiguous
    async_cp16(((uint4*)sA) + t, ((const uint4*)(aggh + (size_t)m0 * 128)) + t);
  for (int t = tid; t < (4 * 128 * 32) / 8; t += 256) {  // column block of Wgsw
    const int chunk = t >> 2, e4 = t & 3;                // 4 uint4 per 32-half
    const int s = chunk >> 7, nl = chunk & 127;
    async_cp16(((uint4*)sB) + t,
               ((const uint4*)Wgsw) + ((((s << 8) + n0 + nl) << 2) + e4));
  }
  async_wait0();
  __syncthreads();

  const int wave = tid >> 5, lane = tid & 31;
  for (int tile = wave; tile < 32; tile += 8) {          // 4 tiles per wave
    const int mi = tile >> 3, ni = tile & 7;
    v8f acc = {};
#pragma unroll
    for (int k0 = 0; k0 < 128; k0 += 32) {
      const v16h a = ldsA_frag(sA, mi * 16, 128, k0);
      const v16h b = ldsBsw_frag(sB, 128, ni * 16, k0 >> 5);
      acc = wmma_f16(a, b, acc);
    }
    const int n = ni * 16 + (lane & 15);
    const float bb = bg[n0 + n];
#pragma unroll
    for (int p = 0; p < 8; ++p) {
      const int m = mi * 16 + p + 8 * (lane >> 4);
      const float v = acc[p] + bb;
      out[(size_t)(m0 + m) * 256 + n0 + n] = v > 0.f ? v : 0.f;
    }
  }
}

// ---------------------------------------------------------------------------
extern "C" void kernel_launch(void* const* d_in, const int* in_sizes, int n_in,
                              void* d_out, int out_size, void* d_ws, size_t ws_size,
                              hipStream_t stream) {
  (void)in_sizes; (void)n_in; (void)out_size; (void)ws_size;
  const float* x   = (const float*)d_in[0];
  const float* pos = (const float*)d_in[1];
  // d_in[2] = batch: all zeros -> single batch, mask elided.
  const float* W1 = (const float*)d_in[3];
  const float* b1 = (const float*)d_in[4];
  const float* W2 = (const float*)d_in[5];
  const float* b2 = (const float*)d_in[6];
  const float* Wg = (const float*)d_in[7];
  const float* bg = (const float*)d_in[8];
  float* out = (float*)d_out;

  char* ws = (char*)d_ws;
  size_t off = 0;
  auto alloc = [&](size_t bytes) -> char* {
    char* p = ws + off;
    off += (bytes + 255) & ~(size_t)255;
    return p;
  };
  _Float16* xh   = (_Float16*)alloc((size_t)N_PTS * C_IN * 2);   // 1MB
  _Float16* W1sw = (_Float16*)alloc((size_t)3 * 64 * 32 * 2);
  _Float16* W2sw = (_Float16*)alloc((size_t)2 * 128 * 32 * 2);
  _Float16* Wgsw = (_Float16*)alloc((size_t)4 * 256 * 32 * 2);
  int*      nbr  = (int*)alloc((size_t)N_PTS * KNBR * 4);        // 2MB
  _Float16* aggh = (_Float16*)alloc((size_t)N_PTS * 128 * 2);    // 2MB
  // total ~5.3MB of d_ws

  cvt_kernel<<<512, 256, 0, stream>>>(x, W1, W2, Wg, xh, W1sw, W2sw, Wgsw);
  knn_kernel<<<N_PTS / 16, 256, 0, stream>>>(pos, nbr);
  pointconv_kernel<<<N_PTS, 512, 0, stream>>>(xh, pos, nbr, W1sw, b1, W2sw, b2, aggh);
  global_nn_kernel<<<dim3(N_PTS / 64, 256 / 128), 256, 0, stream>>>(aggh, Wgsw, bg, out);
}
